// IcaSubspaceModel_18073222382222
// MI455X (gfx1250) — matrix operand order repaired and checked by reading, not compiled, not run on hardware
//
#include <hip/hip_runtime.h>
#include <hip/hip_bf16.h>

typedef __bf16 bf16_t;
typedef __attribute__((ext_vector_type(16))) __bf16 v16bf;
typedef __attribute__((ext_vector_type(8)))  __bf16 v8bf;
typedef __attribute__((ext_vector_type(4)))  __bf16 v4bf;
typedef __attribute__((ext_vector_type(8)))  float   v8f;
typedef __attribute__((ext_vector_type(4)))  float   v4f;

#define LDS_K      32
#define LDS_STRIDE 40     // bf16 elems/row: 32 + 8 pad -> 80B rows, 16B aligned
#define BLK_M      128
#define BLK_N      128

// ---------------- WMMA core: D = A*B + C, bf16 -> f32 ----------------
__device__ __forceinline__ v8f wmma_bf16(v16bf a, v16bf b, v8f c) {
  // (neg_a, A, neg_b, B, c_mod, C, reuse_a, reuse_b)
  return __builtin_amdgcn_wmma_f32_16x16x32_bf16(false, a, false, b, (short)0, c,
                                                 false, false);
}

// A fragment (16x32, 16-bit): lanes 0-15 -> K {0..7,16..23}, lanes 16-31 -> K {8..15,24..31}
__device__ __forceinline__ v16bf frag_a(const bf16_t* __restrict__ row, int lanehi) {
  const int kb = lanehi ? 8 : 0;
  v8bf a = *(const v8bf*)(row + kb);        // K = kb .. kb+7
  v8bf b = *(const v8bf*)(row + kb + 16);   // K = kb+16 .. kb+23
  return __builtin_shufflevector(a, b, 0,1,2,3,4,5,6,7,8,9,10,11,12,13,14,15);
}

// B fragment (32x16, 16-bit): lanes 0-15 -> K {0..7,16..23}, lanes 16-31 -> K {8..15,24..31}
__device__ __forceinline__ v16bf frag_b(const bf16_t* __restrict__ row, int lanehi) {
  const int kb = lanehi ? 8 : 0;
  v8bf a = *(const v8bf*)(row + kb);        // K = kb .. kb+7
  v8bf b = *(const v8bf*)(row + kb + 16);   // K = kb+16 .. kb+23
  return __builtin_shufflevector(a, b, 0,1,2,3,4,5,6,7,8,9,10,11,12,13,14,15);
}

// ---- two-phase tile stage: global fp32 -> regs (fetch), regs -> bf16 hi/lo LDS (commit)
// Logical tile: 128 "rows" x 32 "k".  TRANS=0: src[r*ld+k]; TRANS=1: src[k*ld+r].
template<int TRANS>
__device__ __forceinline__ void fetch_tile(const float* __restrict__ src, int ld,
                                           int tid, v4f (&v)[4]) {
#pragma unroll
  for (int t = 0; t < 4; ++t) {
    const int idx = tid + t * 256;                  // 0..1023 quads
    if (TRANS == 0) {
      const int r = idx >> 3, k4 = (idx & 7) << 2;  // 128B rows, coalesced
      v[t] = *(const v4f*)(src + (long long)r * ld + k4);
    } else {
      const int k = idx >> 5, r4 = (idx & 31) << 2; // 512B rows, coalesced
      v[t] = *(const v4f*)(src + (long long)k * ld + r4);
    }
  }
}

template<int TRANS>
__device__ __forceinline__ void commit_tile(int tid, const v4f (&v)[4],
                                            bf16_t* __restrict__ H,
                                            bf16_t* __restrict__ L) {
#pragma unroll
  for (int t = 0; t < 4; ++t) {
    const int idx = tid + t * 256;
    if (TRANS == 0) {
      const int r = idx >> 3, k4 = (idx & 7) << 2;
      v4bf h, l;
#pragma unroll
      for (int e = 0; e < 4; ++e) {
        const bf16_t hh = (bf16_t)v[t][e];
        h[e] = hh;
        l[e] = (bf16_t)(v[t][e] - (float)hh);
      }
      *(v4bf*)(H + r * LDS_STRIDE + k4) = h;        // ds_store_b64
      *(v4bf*)(L + r * LDS_STRIDE + k4) = l;
    } else {
      const int k = idx >> 5, r4 = (idx & 31) << 2;
#pragma unroll
      for (int e = 0; e < 4; ++e) {                 // transpose: scattered b16 stores
        const bf16_t hh = (bf16_t)v[t][e];
        H[(r4 + e) * LDS_STRIDE + k] = hh;
        L[(r4 + e) * LDS_STRIDE + k] = (bf16_t)(v[t][e] - (float)hh);
      }
    }
  }
}

// ---------------- Generic bf16x3 GEMM: C[m,n] = sum_k A(m,k)*B(n,k) ----------------
// EPI 0: plain store.  EPI 1: store C=s and T = s * (-0.5*rsqrt(group16 sum s^2)).
// EPI 2: store unscaled split-K partial at z-offset.
template<int TRANS_A, int TRANS_B, int EPI>
__global__ void __launch_bounds__(256)
ica_gemm(const float* __restrict__ A, int lda,
         const float* __restrict__ B, int ldb,
         float* __restrict__ C, int ldc,
         int kchunk, long long c_z_stride,
         float* __restrict__ T) {
  // double-buffered tiles: 4 planes x 2 buffers x 10KB = 80KB
  __shared__ __align__(16) bf16_t AHs[2][BLK_M * LDS_STRIDE];
  __shared__ __align__(16) bf16_t ALs[2][BLK_M * LDS_STRIDE];
  __shared__ __align__(16) bf16_t BHs[2][BLK_N * LDS_STRIDE];
  __shared__ __align__(16) bf16_t BLs[2][BLK_N * LDS_STRIDE];

  const int tid  = threadIdx.x;
  const int lane = tid & 31;
  const int wid  = tid >> 5;
  const int wm   = (wid & 3) * 32;    // wave row offset in block (4 waves down)
  const int wn   = (wid >> 2) * 64;   // wave col offset in block (2 waves across)
  const int ml   = lane & 15;
  const int lh   = lane >> 4;

  const long long m0 = (long long)blockIdx.x * BLK_M;
  const long long n0 = (long long)blockIdx.y * BLK_N;
  const long long ks = (long long)blockIdx.z * kchunk;

  v8f acc[2][4];
#pragma unroll
  for (int i = 0; i < 2; ++i)
#pragma unroll
    for (int j = 0; j < 4; ++j)
#pragma unroll
      for (int e = 0; e < 8; ++e) acc[i][j][e] = 0.0f;

  v4f pa[4], pb[4];
  // prologue: stage tile 0 into buffer 0
  fetch_tile<TRANS_A>(TRANS_A ? (A + ks * lda + m0) : (A + m0 * lda + ks), lda, tid, pa);
  fetch_tile<TRANS_B>(TRANS_B ? (B + ks * ldb + n0) : (B + n0 * ldb + ks), ldb, tid, pb);
  commit_tile<TRANS_A>(tid, pa, AHs[0], ALs[0]);
  commit_tile<TRANS_B>(tid, pb, BHs[0], BLs[0]);
  __syncthreads();

  int buf = 0;
  for (int k0 = 0; k0 < kchunk; k0 += LDS_K) {
    const bool last = (k0 + LDS_K >= kchunk);
    if (!last) {
      // issue next tile's global loads before touching the matrix pipes
      const long long kg = ks + k0 + LDS_K;
      fetch_tile<TRANS_A>(TRANS_A ? (A + kg * lda + m0) : (A + m0 * lda + kg), lda, tid, pa);
      fetch_tile<TRANS_B>(TRANS_B ? (B + kg * ldb + n0) : (B + n0 * ldb + kg), ldb, tid, pb);
    }

    const bf16_t* __restrict__ AH = AHs[buf];
    const bf16_t* __restrict__ AL = ALs[buf];
    const bf16_t* __restrict__ BH = BHs[buf];
    const bf16_t* __restrict__ BL = BLs[buf];

    v16bf aH[2], aL[2];
#pragma unroll
    for (int i = 0; i < 2; ++i) {
      const int row = (wm + i * 16 + ml) * LDS_STRIDE;
      aH[i] = frag_a(AH + row, lh);
      aL[i] = frag_a(AL + row, lh);
    }
#pragma unroll
    for (int j = 0; j < 4; ++j) {
      const int row = (wn + j * 16 + ml) * LDS_STRIDE;
      const v16bf bH = frag_b(BH + row, lh);
      const v16bf bL = frag_b(BL + row, lh);
#pragma unroll
      for (int i = 0; i < 2; ++i) {
        acc[i][j] = wmma_bf16(aH[i], bH, acc[i][j]);  // hi*hi
        acc[i][j] = wmma_bf16(aH[i], bL, acc[i][j]);  // hi*lo
        acc[i][j] = wmma_bf16(aL[i], bH, acc[i][j]);  // lo*hi
      }
    }

    if (!last) {
      __syncthreads();                 // all waves done reading buf^1 (prev iter)
      commit_tile<TRANS_A>(tid, pa, AHs[buf ^ 1], ALs[buf ^ 1]);
      commit_tile<TRANS_B>(tid, pb, BHs[buf ^ 1], BLs[buf ^ 1]);
      __syncthreads();                 // next tile visible to all waves
      buf ^= 1;
    }
  }

  float* __restrict__ Cz = C + (long long)blockIdx.z * c_z_stride;
#pragma unroll
  for (int i = 0; i < 2; ++i) {
#pragma unroll
    for (int j = 0; j < 4; ++j) {
      const long long mb = m0 + wm + i * 16 + 8 * lh;  // VGPR e -> row mb+e
      const long long nb = n0 + wn + j * 16 + ml;      // lane   -> column
#pragma unroll
      for (int e = 0; e < 8; ++e) {
        const float v = acc[i][j][e];
        const long long off = (mb + e) * (long long)ldc + nb;
        if (EPI == 1) {
          // group energy: this fragment's 16 columns == exactly one neuron group
          float sq = v * v;
          sq += __shfl_xor(sq, 1);
          sq += __shfl_xor(sq, 2);
          sq += __shfl_xor(sq, 4);
          sq += __shfl_xor(sq, 8);   // reduced within each 16-lane half (fixed row)
          const float nl = -0.5f * rsqrtf(sq);
          Cz[off] = v;        // s
          T[off]  = v * nl;   // s * nl_b (group-broadcast implicit)
        } else {
          Cz[off] = v;        // recon, or split-K partial of X^T * T
        }
      }
    }
  }
}

__global__ void __launch_bounds__(256)
reduce_grad(const float* __restrict__ part, float* __restrict__ out,
            int n, int nz, long long zstride, float scale) {
  const int i = blockIdx.x * blockDim.x + threadIdx.x;
  if (i >= n) return;
  float s = 0.0f;
  for (int z = 0; z < nz; ++z) s += part[(long long)z * zstride + i];
  out[i] = s * scale;
}

extern "C" void kernel_launch(void* const* d_in, const int* in_sizes, int n_in,
                              void* d_out, int out_size, void* d_ws, size_t ws_size,
                              hipStream_t stream) {
  (void)in_sizes; (void)n_in; (void)out_size; (void)ws_size;
  const float* x = (const float*)d_in[0];   // [16384,1024]
  const float* w = (const float*)d_in[1];   // [1024,1024] row-major [N,P]
  // d_in[2] (R) unused: group structure n/16 aligns exactly with 16-wide WMMA tiles.

  const long long BN = 16384LL * 1024LL;
  float* out_s     = (float*)d_out;          // s      [16384,1024]
  float* out_recon = out_s + BN;             // recon  [16384,1024]
  float* out_grad  = out_s + 2 * BN;         // grad   [1024,1024]

  float* T    = (float*)d_ws;                // s*nl_b [16384,1024] fp32 (64 MB)
  float* part = T + BN;                      // 8 x [1024,1024] split-K partials (32 MB)

  const dim3 blk(256, 1, 1);

  // s = X * W^T : A=X (row-major, K contiguous), B[n][k]=W[n][k] (row-major)
  ica_gemm<0, 0, 1><<<dim3(128, 8, 1), blk, 0, stream>>>(
      x, 1024, w, 1024, out_s, 1024, 1024, 0, T);

  // recon = s * W : A=s, B[p][k]=W[k][p] (transposed access)
  ica_gemm<0, 1, 0><<<dim3(128, 8, 1), blk, 0, stream>>>(
      out_s, 1024, w, 1024, out_recon, 1024, 1024, 0, nullptr);

  // grad partials: A[p][b]=X[b][p], B[n][b]=T[b][n], split-K = 8 over batch 16384
  ica_gemm<1, 1, 2><<<dim3(8, 8, 8), blk, 0, stream>>>(
      x, 1024, T, 1024, part, 1024, 2048, 1024LL * 1024LL, nullptr);

  // avg_grad = sum_z partial / 16384
  reduce_grad<<<dim3(4096, 1, 1), blk, 0, stream>>>(
      part, out_grad, 1024 * 1024, 8, 1024LL * 1024LL, 1.0f / 16384.0f);
}
